// FullyConnectSpanRepr_69750268887129
// MI455X (gfx1250) — matrix operand order repaired and checked by reading, not compiled, not run on hardware
//
#include <hip/hip_runtime.h>
#include <hip/hip_bf16.h>

// ---------------------------------------------------------------------------
// FullyConnectSpanRepr on MI455X (gfx1250, wave32, WMMA)
//   BSZ=4 SEQ=48 D=256 NHEAD=2 NLAYER=2 DFF=2048 NQ=512, T = 48*48 = 2304
// All GEMMs run on v_wmma_f32_16x16x32_bf16 (bf16 inputs, f32 accumulate).
// GEMM inner loop is register double-buffered so fragment loads for step
// k+32 overlap the WMMAs of step k (avoids s_wait_loadcnt 0 per WMMA).
// Attention is flash-style (online softmax); fragment loads are batched
// ahead of the WMMA bursts for the same reason.
// ---------------------------------------------------------------------------

typedef unsigned short u16;
typedef unsigned int   u32;
typedef __attribute__((ext_vector_type(16))) __bf16 v16bf;
typedef __attribute__((ext_vector_type(8)))  float  v8f;

#define T_TOK   2304      // SEQ*SEQ
#define MROWS   9216      // BSZ*T_TOK
#define DMODEL  256
#define DHEAD   128
#define SCALE_ATTN 0.08838834764831845f   // 1/sqrt(128)

union Frag16 {
    v16bf v;
    uint4 q[2];
};

__device__ __forceinline__ u16 f2bf(float f) {
    u32 u = __builtin_bit_cast(u32, f);
    u += 0x7fffu + ((u >> 16) & 1u);      // round-to-nearest-even
    return (u16)(u >> 16);
}

// ---------------------------------------------------------------------------
// f32 -> bf16 bulk convert (weights)
// ---------------------------------------------------------------------------
__global__ __launch_bounds__(256) void cvt_bf16_kernel(const float* __restrict__ in,
                                                       u16* __restrict__ out, int n) {
    int i = blockIdx.x * 256 + threadIdx.x;
    if (i < n) out[i] = f2bf(in[i]);
}

// ---------------------------------------------------------------------------
// span_max: X[b,s,e,:] = max(enc[b, s..e, :]) (0 when e < s); also bf16 copy.
// grid (48, 4), block 256 (one thread per dim)
// ---------------------------------------------------------------------------
__global__ __launch_bounds__(256) void span_kernel(const float* __restrict__ enc,
                                                   float* __restrict__ X,
                                                   u16* __restrict__ XB) {
    int s = blockIdx.x, b = blockIdx.y, d = threadIdx.x;
    float running = 0.0f;
    for (int e = 0; e < 48; ++e) {
        float val;
        if (e < s) {
            val = 0.0f;
        } else {
            float x = enc[((size_t)(b * 48 + e)) * DMODEL + d];
            running = (e == s) ? x : fmaxf(running, x);
            val = running;
        }
        size_t row = (size_t)(b * 48 + s) * 48 + e;
        X [row * DMODEL + d] = val;
        XB[row * DMODEL + d] = f2bf(val);
    }
}

// ---------------------------------------------------------------------------
// Generic GEMM: C(M x N) = A(M x K, bf16) * W(N x K, bf16)^T + bias
//   each wave computes a 16(M) x 64(N) tile; grid (N/64, M/128), block 256.
//   Register double-buffered over K (K % 64 == 0 at all call sites).
// ---------------------------------------------------------------------------
__global__ __launch_bounds__(256) void gemm_bf16_kernel(const u16* __restrict__ A,
                                                        const u16* __restrict__ W,
                                                        const float* __restrict__ bias,
                                                        float* __restrict__ outF,
                                                        u16* __restrict__ outB,
                                                        int K, int N, int relu) {
    const int lane = threadIdx.x & 31;
    const int wave = threadIdx.x >> 5;
    const int l16  = lane & 15;
    const int lh   = lane >> 4;

    const int m0 = (blockIdx.y * 8 + wave) * 16;
    const int n0 = blockIdx.x * 64;

    v8f c[4];
#pragma unroll
    for (int j = 0; j < 4; ++j)
#pragma unroll
        for (int v = 0; v < 8; ++v) c[j][v] = 0.0f;

    const u16* ap  = A + (size_t)(m0 + l16) * K;
    const u16* wp0 = W + (size_t)(n0 + l16) * K;

    Frag16 aP, aN;
    Frag16 bP[4], bN[4];

    auto loadA = [&](Frag16& f, int kb) {
        f.q[0] = *(const uint4*)(ap + kb + lh * 8);
        f.q[1] = *(const uint4*)(ap + kb + 16 + lh * 8);
    };
    auto loadB = [&](Frag16* f, int kb) {
#pragma unroll
        for (int j = 0; j < 4; ++j) {
            const u16* wp = wp0 + (size_t)(j * 16) * K + kb + lh * 8;
            f[j].q[0] = *(const uint4*)(wp);
            f[j].q[1] = *(const uint4*)(wp + 16);
        }
    };

    loadA(aP, 0);
    loadB(bP, 0);

    for (int kb = 0; kb < K; kb += 64) {
        // issue loads for the odd half-step before consuming the even one
        loadA(aN, kb + 32);
        loadB(bN, kb + 32);
        __builtin_prefetch(ap + kb + 256, 0, 1);
#pragma unroll
        for (int j = 0; j < 4; ++j)
            c[j] = __builtin_amdgcn_wmma_f32_16x16x32_bf16(
                false, aP.v, false, bP[j].v, (short)0, c[j], false, false);

        if (kb + 64 < K) {                    // uniform scalar branch
            loadA(aP, kb + 64);
            loadB(bP, kb + 64);
        }
#pragma unroll
        for (int j = 0; j < 4; ++j)
            c[j] = __builtin_amdgcn_wmma_f32_16x16x32_bf16(
                false, aN.v, false, bN[j].v, (short)0, c[j], false, false);
    }

#pragma unroll
    for (int j = 0; j < 4; ++j) {
        const int n = n0 + j * 16 + l16;
        const float bb = bias ? bias[n] : 0.0f;
#pragma unroll
        for (int v = 0; v < 8; ++v) {
            const int m = m0 + lh * 8 + v;
            float val = c[j][v] + bb;
            if (relu) val = fmaxf(val, 0.0f);
            if (outF) outF[(size_t)m * N + n] = val;
            if (outB) outB[(size_t)m * N + n] = f2bf(val);
        }
    }
}

// ---------------------------------------------------------------------------
// Build V^T: VT[bh][d][t] = qkvB[b*T+t][512 + h*128 + d]
// ---------------------------------------------------------------------------
__global__ __launch_bounds__(256) void vt_kernel(const u16* __restrict__ qkvB,
                                                 u16* __restrict__ VT) {
    int idx = blockIdx.x * 256 + threadIdx.x;              // 8*128*2304 total
    int bh  = idx / (DHEAD * T_TOK);
    int rem = idx % (DHEAD * T_TOK);
    int d   = rem / T_TOK;
    int t   = rem % T_TOK;
    int b = bh >> 1, h = bh & 1;
    VT[idx] = qkvB[(size_t)(b * T_TOK + t) * 768 + 512 + h * DHEAD + d];
}

// ---------------------------------------------------------------------------
// Flash attention, one wave per 16-query tile.  grid (144, 8), block 32.
//   Key mask depends only on the key token (e < s -> -1e9), same for all q.
// ---------------------------------------------------------------------------
__global__ __launch_bounds__(32) void attn_kernel(const u16* __restrict__ qkvB,
                                                  const u16* __restrict__ VT,
                                                  u16* __restrict__ attnB) {
    __shared__ __align__(16) u16 Plds[16][40];   // 16 q-rows x 32 keys (+pad)

    const int lane  = threadIdx.x;
    const int qtile = blockIdx.x;                // 0..143
    const int bh    = blockIdx.y;                // 0..7
    const int b = bh >> 1, h = bh & 1;
    const int l16 = lane & 15;
    const int lh  = lane >> 4;

    const u16* Q  = qkvB + (size_t)(b * T_TOK) * 768 + h * DHEAD;
    const u16* Kp = Q + 256;
    const u16* Vt = VT + (size_t)bh * DHEAD * T_TOK;

    // Q fragments for the whole tile (head dim 128 = 4 WMMA k-steps)
    Frag16 qf[4];
    {
        const u16* qp = Q + (size_t)(qtile * 16 + l16) * 768;
#pragma unroll
        for (int kd = 0; kd < 4; ++kd) {
            qf[kd].q[0] = *(const uint4*)(qp + kd * 32 + lh * 8);
            qf[kd].q[1] = *(const uint4*)(qp + kd * 32 + 16 + lh * 8);
        }
    }

    float m[8], l[8];
    v8f o[8];
#pragma unroll
    for (int v = 0; v < 8; ++v) { m[v] = -3.0e38f; l[v] = 0.0f; }
#pragma unroll
    for (int j = 0; j < 8; ++j)
#pragma unroll
        for (int v = 0; v < 8; ++v) o[j][v] = 0.0f;

    for (int kt = 0; kt < T_TOK / 32; ++kt) {
        const int kb = kt * 32;
        const int key0 = kb + l16;
        const int key1 = kb + 16 + l16;
        const float mb0 = ((key0 % 48) < (key0 / 48)) ? -1.0e9f : 0.0f;
        const float mb1 = ((key1 % 48) < (key1 / 48)) ? -1.0e9f : 0.0f;

        v8f s0, s1;
#pragma unroll
        for (int v = 0; v < 8; ++v) { s0[v] = 0.0f; s1[v] = 0.0f; }

        // batch all 8 K-fragment loads (one clause / one wait) ...
        Frag16 kf0[4], kf1[4];
#pragma unroll
        for (int kd = 0; kd < 4; ++kd) {
            const u16* kp0 = Kp + (size_t)key0 * 768 + kd * 32 + lh * 8;
            const u16* kp1 = Kp + (size_t)key1 * 768 + kd * 32 + lh * 8;
            kf0[kd].q[0] = *(const uint4*)(kp0);
            kf0[kd].q[1] = *(const uint4*)(kp0 + 16);
            kf1[kd].q[0] = *(const uint4*)(kp1);
            kf1[kd].q[1] = *(const uint4*)(kp1 + 16);
        }
        // ... then run the 8 QK^T WMMAs back-to-back
#pragma unroll
        for (int kd = 0; kd < 4; ++kd) {
            s0 = __builtin_amdgcn_wmma_f32_16x16x32_bf16(
                false, qf[kd].v, false, kf0[kd].v, (short)0, s0, false, false);
            s1 = __builtin_amdgcn_wmma_f32_16x16x32_bf16(
                false, qf[kd].v, false, kf1[kd].v, (short)0, s1, false, false);
        }

        // online softmax update (row = (lh*8+v), spread over 16-lane group)
#pragma unroll
        for (int v = 0; v < 8; ++v) {
            float a0 = s0[v] * SCALE_ATTN + mb0;
            float a1 = s1[v] * SCALE_ATTN + mb1;
            float cmax = fmaxf(a0, a1);
            cmax = fmaxf(cmax, __shfl_xor(cmax, 1, 32));
            cmax = fmaxf(cmax, __shfl_xor(cmax, 2, 32));
            cmax = fmaxf(cmax, __shfl_xor(cmax, 4, 32));
            cmax = fmaxf(cmax, __shfl_xor(cmax, 8, 32));
            float mn = fmaxf(m[v], cmax);
            float sc = __expf(m[v] - mn);
            float p0 = __expf(a0 - mn);
            float p1 = __expf(a1 - mn);
            float r = p0 + p1;
            r += __shfl_xor(r, 1, 32);
            r += __shfl_xor(r, 2, 32);
            r += __shfl_xor(r, 4, 32);
            r += __shfl_xor(r, 8, 32);
            l[v] = l[v] * sc + r;
            m[v] = mn;
#pragma unroll
            for (int j = 0; j < 8; ++j) o[j][v] *= sc;
            Plds[lh * 8 + v][l16]      = f2bf(p0);
            Plds[lh * 8 + v][16 + l16] = f2bf(p1);
        }
        __syncthreads();

        // P A-fragment (16 q x 32 key) straight from LDS in layout order
        Frag16 pf;
        {
            const u16* pr = &Plds[l16][lh * 8];
            pf.q[0] = *(const uint4*)(pr);
            pf.q[1] = *(const uint4*)(pr + 16);
        }

        // batch all 8 V-fragment loads, then 8 PV WMMAs back-to-back
        Frag16 vf[8];
#pragma unroll
        for (int j = 0; j < 8; ++j) {
            const u16* vp = Vt + (size_t)(j * 16 + l16) * T_TOK + kb + lh * 8;
            vf[j].q[0] = *(const uint4*)(vp);
            vf[j].q[1] = *(const uint4*)(vp + 16);
        }
#pragma unroll
        for (int j = 0; j < 8; ++j)
            o[j] = __builtin_amdgcn_wmma_f32_16x16x32_bf16(
                false, pf.v, false, vf[j].v, (short)0, o[j], false, false);
        __syncthreads();
    }

    // epilogue: divide by row sum, store bf16 into attnB (concat-head layout)
    const int orow = b * T_TOK + qtile * 16 + lh * 8;
#pragma unroll
    for (int j = 0; j < 8; ++j) {
        const int col = h * DHEAD + j * 16 + l16;
#pragma unroll
        for (int v = 0; v < 8; ++v) {
            float val = o[j][v] / l[v];
            attnB[(size_t)(orow + v) * DMODEL + col] = f2bf(val);
        }
    }
}

// ---------------------------------------------------------------------------
// LayerNorm (+ optional residual add), per row of 256.  grid 9216, block 256.
// ---------------------------------------------------------------------------
__global__ __launch_bounds__(256) void ln_kernel(const float* __restrict__ Xin,
                                                 const float* __restrict__ Gin,
                                                 const float* __restrict__ w,
                                                 const float* __restrict__ bvec,
                                                 float* __restrict__ outF,
                                                 u16* __restrict__ outB) {
    __shared__ float red[256];
    const int row = blockIdx.x, d = threadIdx.x;
    float r = Xin[(size_t)row * DMODEL + d];
    if (Gin) r += Gin[(size_t)row * DMODEL + d];

    red[d] = r; __syncthreads();
    for (int s = 128; s > 0; s >>= 1) {
        if (d < s) red[d] += red[d + s];
        __syncthreads();
    }
    const float mean = red[0] * (1.0f / DMODEL);
    __syncthreads();

    const float df = r - mean;
    red[d] = df * df; __syncthreads();
    for (int s = 128; s > 0; s >>= 1) {
        if (d < s) red[d] += red[d + s];
        __syncthreads();
    }
    const float var = red[0] * (1.0f / DMODEL);

    const float y = df * rsqrtf(var + 1e-5f) * w[d] + bvec[d];
    if (outF) outF[(size_t)row * DMODEL + d] = y;
    if (outB) outB[(size_t)row * DMODEL + d] = f2bf(y);
}

// ---------------------------------------------------------------------------
// Gather res1/res2: out[q] = F[qb, s, e].  grid 1024, block 256.
// ---------------------------------------------------------------------------
__global__ __launch_bounds__(256) void gather_kernel(const float* __restrict__ F,
                                                     const int* __restrict__ s1,
                                                     const int* __restrict__ e1,
                                                     const int* __restrict__ qb,
                                                     const int* __restrict__ s2,
                                                     const int* __restrict__ e2,
                                                     float* __restrict__ out) {
    const int q = blockIdx.x, d = threadIdx.x;
    int i, s, e;
    if (q < 512) { i = q;        s = s1[i]; e = e1[i]; }
    else         { i = q - 512;  s = s2[i]; e = e2[i]; }
    const int b = qb[i];
    const size_t row = (size_t)b * T_TOK + (size_t)s * 48 + e;
    out[(size_t)q * DMODEL + d] = F[row * DMODEL + d];
}

// ---------------------------------------------------------------------------
// Host driver
// ---------------------------------------------------------------------------
extern "C" void kernel_launch(void* const* d_in, const int* in_sizes, int n_in,
                              void* d_out, int out_size, void* d_ws, size_t ws_size,
                              hipStream_t stream) {
    (void)in_sizes; (void)n_in; (void)out_size; (void)ws_size;

    const float* enc = (const float*)d_in[0];
    const float* iw  = (const float*)d_in[1];
    const float* ibv = (const float*)d_in[2];
    const float* ow  = (const float*)d_in[3];
    const float* obv = (const float*)d_in[4];
    const float* l1w = (const float*)d_in[5];
    const float* l1b = (const float*)d_in[6];
    const float* l2w = (const float*)d_in[7];
    const float* l2b = (const float*)d_in[8];
    const float* f1w = (const float*)d_in[9];
    const float* f1b = (const float*)d_in[10];
    const float* f2w = (const float*)d_in[11];
    const float* f2b = (const float*)d_in[12];
    const float* flw = (const float*)d_in[13];
    const float* flb = (const float*)d_in[14];
    const int*   s1  = (const int*)d_in[15];
    const int*   e1  = (const int*)d_in[16];
    const int*   qb  = (const int*)d_in[17];
    const int*   s2  = (const int*)d_in[18];
    const int*   e2  = (const int*)d_in[19];

    // workspace carve-up (256B aligned), total ~91 MB
    char* p = (char*)d_ws;
    auto alloc = [&](size_t bytes) -> void* {
        void* r = (void*)p;
        p += (bytes + 255) & ~(size_t)255;
        return r;
    };
    float* X     = (float*)alloc((size_t)MROWS * DMODEL * 4);   // hidden f32
    float* G     = (float*)alloc((size_t)MROWS * DMODEL * 4);   // gemm f32 out
    u16*   XB    = (u16*)  alloc((size_t)MROWS * DMODEL * 2);   // hidden bf16
    u16*   qkvB  = (u16*)  alloc((size_t)MROWS * 768    * 2);
    u16*   VTb   = (u16*)  alloc((size_t)8 * DHEAD * T_TOK * 2);
    u16*   attnB = (u16*)  alloc((size_t)MROWS * DMODEL * 2);
    u16*   Hb    = (u16*)  alloc((size_t)MROWS * 2048   * 2);   // FF hidden bf16
    u16*   iwB   = (u16*)  alloc((size_t)2 * 768  * 256  * 2);
    u16*   owB   = (u16*)  alloc((size_t)2 * 256  * 256  * 2);
    u16*   f1B   = (u16*)  alloc((size_t)2 * 2048 * 256  * 2);
    u16*   f2B   = (u16*)  alloc((size_t)2 * 256  * 2048 * 2);

    // weight conversion to bf16
    auto cvt = [&](const float* src, u16* dst, int n) {
        cvt_bf16_kernel<<<(n + 255) / 256, 256, 0, stream>>>(src, dst, n);
    };
    cvt(iw,  iwB, 2 * 768 * 256);
    cvt(ow,  owB, 2 * 256 * 256);
    cvt(f1w, f1B, 2 * 2048 * 256);
    cvt(f2w, f2B, 2 * 256 * 2048);

    // span-max expansion: (4,48,256) -> (9216,256) f32 + bf16
    span_kernel<<<dim3(48, 4), 256, 0, stream>>>(enc, X, XB);

    for (int l = 0; l < 2; ++l) {
        // QKV projection -> bf16 qkv buffer (M=9216, K=256, N=768)
        gemm_bf16_kernel<<<dim3(768 / 64, MROWS / 128), 256, 0, stream>>>(
            XB, iwB + (size_t)l * 768 * 256, ibv + (size_t)l * 768,
            nullptr, qkvB, 256, 768, 0);

        // V^T for contiguous B-fragments in PV
        vt_kernel<<<(8 * DHEAD * T_TOK) / 256, 256, 0, stream>>>(qkvB, VTb);

        // flash attention -> attnB bf16 (concat-head)
        attn_kernel<<<dim3(T_TOK / 16, 8), 32, 0, stream>>>(qkvB, VTb, attnB);

        // out projection -> G f32 (M=9216, K=256, N=256)
        gemm_bf16_kernel<<<dim3(256 / 64, MROWS / 128), 256, 0, stream>>>(
            attnB, owB + (size_t)l * 256 * 256, obv + (size_t)l * 256,
            G, nullptr, 256, 256, 0);

        // x = LN(x + attn)
        ln_kernel<<<MROWS, 256, 0, stream>>>(X, G, l1w + l * 256, l1b + l * 256, X, XB);

        // FF1 with ReLU -> Hb bf16 (M=9216, K=256, N=2048)
        gemm_bf16_kernel<<<dim3(2048 / 64, MROWS / 128), 256, 0, stream>>>(
            XB, f1B + (size_t)l * 2048 * 256, f1b + (size_t)l * 2048,
            nullptr, Hb, 256, 2048, 1);

        // FF2 -> G f32 (M=9216, K=2048, N=256)
        gemm_bf16_kernel<<<dim3(256 / 64, MROWS / 128), 256, 0, stream>>>(
            Hb, f2B + (size_t)l * 256 * 2048, f2b + (size_t)l * 256,
            G, nullptr, 2048, 256, 0);

        // x = LN(x + ff)
        ln_kernel<<<MROWS, 256, 0, stream>>>(X, G, l2w + l * 256, l2b + l * 256, X, XB);
    }

    // final LN -> G (f32 only)
    ln_kernel<<<MROWS, 256, 0, stream>>>(X, nullptr, flw, flb, G, nullptr);

    // gather res1 | res2 into d_out
    gather_kernel<<<1024, 256, 0, stream>>>(G, s1, e1, qb, s2, e2, (float*)d_out);
}
